// NodeAttentionPerMetaPath_1159641170254
// MI455X (gfx1250) — matrix operand order repaired and checked by reading, not compiled, not run on hardware
//
#include <hip/hip_runtime.h>
#include <hip/hip_bf16.h>

typedef __attribute__((ext_vector_type(2)))  float  v2f;
typedef __attribute__((ext_vector_type(8)))  float  v8f;
typedef __attribute__((ext_vector_type(16))) __bf16 v16bf;
typedef __attribute__((ext_vector_type(2)))  __bf16 v2bf;
typedef __attribute__((ext_vector_type(4)))  int    v4i;

#define N_NODES 8192
#define F_IN    256
#define D_OUT   64
#define NEG_INF (-1.0e15f)
#define ALPHA   0.2f

// Pack two floats into one dword of bf16 (lo -> [15:0], hi -> [31:16]).
// gfx1250 is a native-bf16 target: let clang lower the fptrunc to the HW cvt.
__device__ __forceinline__ unsigned int pack_bf16x2(float lo, float hi) {
    v2bf v;
    v.x = (__bf16)lo;
    v.y = (__bf16)hi;
    return __builtin_bit_cast(unsigned int, v);
}

union BfFrag {
    unsigned int   u[8];
    uint4          q[2];
    v16bf          v;
};

// ---------------------------------------------------------------------------
// Kernel 1: h = x @ trans via V_WMMA_F32_16X16X4_F32 (exact fp32 path),
// e1 = h@attn[:64], e2 = h@attn[64:]. Stores hT bf16 [D_OUT][N_NODES].
// One wave -> 16x64 tile (4 accumulators).
// ---------------------------------------------------------------------------
__global__ void __launch_bounds__(256)
k_proj(const float* __restrict__ x, const float* __restrict__ trans,
       const float* __restrict__ attn,
       unsigned short* __restrict__ hT,
       float* __restrict__ e1, float* __restrict__ e2)
{
    const int wave = (blockIdx.x * blockDim.x + threadIdx.x) >> 5;
    const int lane = threadIdx.x & 31;
    const int lo = lane & 15;      // A row / C column within tile
    const int hi = lane >> 4;      // half-wave select
    const int r0 = wave * 16;

    v8f acc[4] = {};
    const float* xrow = x + (size_t)(r0 + lo) * F_IN;

    for (int kb = 0; kb < F_IN; kb += 4) {
        // A 16x4 fp32 frag: lanes0-15 -> K=kb,kb+1 ; lanes16-31 -> K=kb+2,kb+3
        v2f a = *(const v2f*)(xrow + kb + hi * 2);
        const int krow = kb + hi * 2;
#pragma unroll
        for (int t = 0; t < 4; ++t) {
            // B 4x16 fp32 frag: lane column = lo; VGPR0 = even K row, VGPR1 = odd
            v2f b;
            b.x = trans[(size_t)krow * D_OUT + t * 16 + lo];
            b.y = trans[(size_t)(krow + 1) * D_OUT + t * 16 + lo];
            acc[t] = __builtin_amdgcn_wmma_f32_16x16x4_f32(
                false, a, false, b, (short)0, acc[t], false, false);
        }
    }

    // C/D layout: lane holds rows M = v + 8*hi, column = t*16 + lo.
    // Rows v=0..7 are contiguous in hT -> pack 8 bf16 into one 16B store.
#pragma unroll
    for (int t = 0; t < 4; ++t) {
        uint4 st;
        st.x = pack_bf16x2(acc[t][0], acc[t][1]);
        st.y = pack_bf16x2(acc[t][2], acc[t][3]);
        st.z = pack_bf16x2(acc[t][4], acc[t][5]);
        st.w = pack_bf16x2(acc[t][6], acc[t][7]);
        *(uint4*)(hT + (size_t)(t * 16 + lo) * N_NODES + r0 + 8 * hi) = st;
    }

    // e1/e2: per-row dot with attn vectors, reduced across 16 lanes of a half.
#pragma unroll
    for (int v = 0; v < 8; ++v) {
        float p1 = 0.f, p2 = 0.f;
#pragma unroll
        for (int t = 0; t < 4; ++t) {
            const float hval = acc[t][v];
            p1 += hval * attn[t * 16 + lo];
            p2 += hval * attn[D_OUT + t * 16 + lo];
        }
#pragma unroll
        for (int mskv = 1; mskv < 16; mskv <<= 1) {   // stays within 16-lane half
            p1 += __shfl_xor(p1, mskv, 32);
            p2 += __shfl_xor(p2, mskv, 32);
        }
        if (lo == 0) {
            e1[r0 + v + 8 * hi] = p1;
            e2[r0 + v + 8 * hi] = p2;
        }
    }
}

// ---------------------------------------------------------------------------
// Kernel 1.5: global max of e2 (gives an exact softmax row-max upper bound,
// since scores[i,j] = leakyrelu(e1[i]+e2[j]) and leakyrelu is monotonic).
// ---------------------------------------------------------------------------
__global__ void __launch_bounds__(256)
k_e2max(const float* __restrict__ e2, float* __restrict__ e2max)
{
    __shared__ float red[256];
    float v = -3.0e38f;
    for (int i = threadIdx.x; i < N_NODES; i += 256)
        v = fmaxf(v, e2[i]);
    red[threadIdx.x] = v;
    __syncthreads();
    for (int s = 128; s > 0; s >>= 1) {
        if (threadIdx.x < (unsigned)s)
            red[threadIdx.x] = fmaxf(red[threadIdx.x], red[threadIdx.x + s]);
        __syncthreads();
    }
    if (threadIdx.x == 0) *e2max = red[0];
}

// ---------------------------------------------------------------------------
// Kernel 2: fused masked-softmax attention, single pass over the mask with a
// FIXED per-row max bound (no online rescale, no per-chunk shuffles).
// P[16x32] @ h[32x64] via V_WMMA_F32_16X16X32_BF16, fp32 accumulators.
// ---------------------------------------------------------------------------
__global__ void __launch_bounds__(256)
k_flash_attn(const int* __restrict__ mask, const unsigned short* __restrict__ hT,
             const float* __restrict__ e1, const float* __restrict__ e2,
             const float* __restrict__ e2max_p, float* __restrict__ out)
{
    const int wave = (blockIdx.x * blockDim.x + threadIdx.x) >> 5;
    const int lane = threadIdx.x & 31;
    const int lo = lane & 15;
    const int hi = lane >> 4;
    const int r0 = wave * 16;
    const int kb0 = hi * 8;    // A-frag K runs: [kb0,kb0+8) and [kb0+16,kb0+24)
    const int kbB = hi * 8;    // B-frag K runs: [kbB,kbB+8) and [kbB+16,kbB+24)

    const float e1v = e1[r0 + lo];
    const float ymax = e1v + *e2max_p;
    const float m_i = fmaxf(ymax, ALPHA * ymax);   // exact row-max upper bound
    float ls0 = 0.f, ls1 = 0.f, ls2 = 0.f, ls3 = 0.f;  // 4 partial sums (ILP)
    v8f acc[4] = {};

    const int* mrow = mask + (size_t)(r0 + lo) * N_NODES;

#pragma unroll 2
    for (int j = 0; j < N_NODES; j += 32) {
        // This lane's 16 mask / e2 values (two contiguous 8-element runs).
        // Mask is 256MB streamed once -> non-temporal so hT/e2 stay in L2.
        v4i m0 = __builtin_nontemporal_load((const v4i*)(mrow + j + kb0));
        v4i m1 = __builtin_nontemporal_load((const v4i*)(mrow + j + kb0 + 4));
        v4i m2 = __builtin_nontemporal_load((const v4i*)(mrow + j + kb0 + 16));
        v4i m3 = __builtin_nontemporal_load((const v4i*)(mrow + j + kb0 + 20));
        float4 f0 = *(const float4*)(e2 + j + kb0);
        float4 f1 = *(const float4*)(e2 + j + kb0 + 4);
        float4 f2 = *(const float4*)(e2 + j + kb0 + 16);
        float4 f3 = *(const float4*)(e2 + j + kb0 + 20);

        const int   mk[16] = { m0.x, m0.y, m0.z, m0.w, m1.x, m1.y, m1.z, m1.w,
                               m2.x, m2.y, m2.z, m2.w, m3.x, m3.y, m3.z, m3.w };
        const float ek[16] = { f0.x, f0.y, f0.z, f0.w, f1.x, f1.y, f1.z, f1.w,
                               f2.x, f2.y, f2.z, f2.w, f3.x, f3.y, f3.z, f3.w };

        float p[16];
#pragma unroll
        for (int i = 0; i < 16; ++i) {
            const float y  = e1v + ek[i];
            const float lr = fmaxf(y, ALPHA * y);            // leaky relu
            const float sc = mk[i] != 0 ? lr : NEG_INF;      // mask fill
            p[i] = __expf(sc - m_i);                         // masked -> exp(-1e15) = 0
        }
#pragma unroll
        for (int i = 0; i < 16; i += 4) {                    // 4-way sum ILP
            ls0 += p[i];
            ls1 += p[i + 1];
            ls2 += p[i + 2];
            ls3 += p[i + 3];
        }

        // Pack P into the bf16 A-fragment (native bf16 cvt on gfx1250).
        BfFrag a;
#pragma unroll
        for (int w = 0; w < 8; ++w)
            a.u[w] = pack_bf16x2(p[2 * w], p[2 * w + 1]);

        // P[16x32] @ h[32x64]: 4 WMMAs; B frag = two 8-bf16 runs (k, k+16) from hT.
#pragma unroll
        for (int t = 0; t < 4; ++t) {
            BfFrag b;
            const unsigned short* hp = hT + (size_t)(t * 16 + lo) * N_NODES + j + kbB;
            b.q[0] = *(const uint4*)(hp);
            b.q[1] = *(const uint4*)(hp + 16);
            acc[t] = __builtin_amdgcn_wmma_f32_16x16x32_bf16(
                false, a.v, false, b.v, (short)0, acc[t], false, false);
        }
    }

    // Combine the two half-row partial sums, normalize, write out fp32.
    float lsum = (ls0 + ls1) + (ls2 + ls3);
    lsum += __shfl_xor(lsum, 16, 32);
    const float rinv = 1.0f / lsum;
    float lv[8];
#pragma unroll
    for (int v = 0; v < 8; ++v) lv[v] = __shfl(rinv, v + 8 * hi, 32);
#pragma unroll
    for (int t = 0; t < 4; ++t)
#pragma unroll
        for (int v = 0; v < 8; ++v)
            out[(size_t)(r0 + v + 8 * hi) * D_OUT + t * 16 + lo] = acc[t][v] * lv[v];
}

extern "C" void kernel_launch(void* const* d_in, const int* in_sizes, int n_in,
                              void* d_out, int out_size, void* d_ws, size_t ws_size,
                              hipStream_t stream) {
    const float* x     = (const float*)d_in[0];
    const int*   mask  = (const int*)d_in[1];
    const float* trans = (const float*)d_in[2];
    const float* attn  = (const float*)d_in[3];
    float* out = (float*)d_out;

    unsigned short* hT = (unsigned short*)d_ws;                         // 1 MB
    float* e1 = (float*)((char*)d_ws + (size_t)D_OUT * N_NODES * 2);    // 32 KB
    float* e2 = e1 + N_NODES;                                           // 32 KB
    float* e2max = e2 + N_NODES;                                        // 4 B

    // 512 waves (one per 16-row tile), 8 waves per block.
    k_proj<<<64, 256, 0, stream>>>(x, trans, attn, hT, e1, e2);
    k_e2max<<<1, 256, 0, stream>>>(e2, e2max);
    k_flash_attn<<<64, 256, 0, stream>>>(mask, hT, e1, e2, e2max, out);
}
